// CausalMultiHeadAttention_88072599372318
// MI455X (gfx1250) — matrix operand order repaired and checked by reading, not compiled, not run on hardware
//
#include <hip/hip_runtime.h>
#include <hip/hip_bf16.h>

// ---------------------------------------------------------------------------
// CDNA5 (gfx1250) wave32 WMMA types
// ---------------------------------------------------------------------------
typedef __attribute__((ext_vector_type(16))) __bf16 v16bf;
typedef __attribute__((ext_vector_type(8)))  __bf16 v8bf;
typedef __attribute__((ext_vector_type(4)))  __bf16 v4bf;
typedef __attribute__((ext_vector_type(8)))  float  v8f;

#define WMMA_BF16(A, B, C) \
  __builtin_amdgcn_wmma_f32_16x16x32_bf16(false, (A), false, (B), (short)0, (C), false, false)

static constexpr int BATCH  = 4;
static constexpr int SEQ    = 2048;
static constexpr int DMODEL = 1024;
static constexpr int NHEADS = 16;
static constexpr int HDIM   = 64;
static constexpr int MROWS  = BATCH * SEQ;     // 8192

__device__ __forceinline__ v16bf cat8(v8bf lo, v8bf hi) {
  return __builtin_shufflevector(lo, hi, 0,1,2,3,4,5,6,7,8,9,10,11,12,13,14,15);
}

// A fragment (16x32 MxK, bf16). src row-major, base = &src[row0*ld + k0].
// Layout: M=lane&15, element s holds K = 16*hi + s (contiguous per half-wave).
__device__ __forceinline__ v16bf load_a(const __bf16* base, int ld, int lane) {
  const int m  = lane & 15;
  const int hi = lane >> 4;
  const __bf16* p = base + m * ld + hi * 16;
  v8bf lo = *(const v8bf*)p;          // K = hi*16 + [0..8)
  v8bf hh = *(const v8bf*)(p + 8);    // K = hi*16 + 8 + [0..8)
  return cat8(lo, hh);
}

// B fragment (32x16 KxN, bf16) where the source stores K contiguously per N
// row: element (k,n) at src[n*ld + k].  base = &src[n0*ld + k0].
// Layout (contiguous, matches load_a): lanes 0-15 hold K=[0..16), lanes 16-31 hold K=[16..32), N=lane&15.
__device__ __forceinline__ v16bf load_bT(const __bf16* base, int ld, int lane) {
  const int n  = lane & 15;
  const int hi = lane >> 4;
  const __bf16* p = base + n * ld + hi * 16;
  v8bf lo = *(const v8bf*)p;
  v8bf hh = *(const v8bf*)(p + 8);
  return cat8(lo, hh);
}

// ---------------------------------------------------------------------------
// Shared GEMM core: one wave computes a 16(M) x 64(N) tile over K=1024.
// K-loop unrolled x2 with two named buffer sets -> one-step load lookahead
// with NO register rotation copies between iterations.
// ---------------------------------------------------------------------------
__device__ __forceinline__ void gemm_16x64(const __bf16* __restrict__ abase,
                                           const __bf16* __restrict__ bbase,
                                           int lane, v8f acc[4]) {
  v16bf a0 = load_a(abase, DMODEL, lane);
  v16bf b0[4];
#pragma unroll
  for (int j = 0; j < 4; ++j)
    b0[j] = load_bT(bbase + (size_t)(j * 16) * DMODEL, DMODEL, lane);

  v16bf a1, b1[4];
  for (int k0 = 0; k0 < DMODEL - 64; k0 += 64) {
    a1 = load_a(abase + k0 + 32, DMODEL, lane);
#pragma unroll
    for (int j = 0; j < 4; ++j)
      b1[j] = load_bT(bbase + (size_t)(j * 16) * DMODEL + k0 + 32, DMODEL, lane);
#pragma unroll
    for (int j = 0; j < 4; ++j)
      acc[j] = WMMA_BF16(a0, b0[j], acc[j]);

    a0 = load_a(abase + k0 + 64, DMODEL, lane);
#pragma unroll
    for (int j = 0; j < 4; ++j)
      b0[j] = load_bT(bbase + (size_t)(j * 16) * DMODEL + k0 + 64, DMODEL, lane);
#pragma unroll
    for (int j = 0; j < 4; ++j)
      acc[j] = WMMA_BF16(a1, b1[j], acc[j]);
  }
  // tail: k0 = DMODEL-64 in buf0, DMODEL-32 in buf1
  a1 = load_a(abase + DMODEL - 32, DMODEL, lane);
#pragma unroll
  for (int j = 0; j < 4; ++j)
    b1[j] = load_bT(bbase + (size_t)(j * 16) * DMODEL + DMODEL - 32, DMODEL, lane);
#pragma unroll
  for (int j = 0; j < 4; ++j)
    acc[j] = WMMA_BF16(a0, b0[j], acc[j]);
#pragma unroll
  for (int j = 0; j < 4; ++j)
    acc[j] = WMMA_BF16(a1, b1[j], acc[j]);
}

// ---------------------------------------------------------------------------
// Stage 0: fp32 -> bf16 conversion, 4 elements / thread
// ---------------------------------------------------------------------------
__global__ void cvt_f32_to_bf16(const float4* __restrict__ src,
                                v4bf* __restrict__ dst, int n4) {
  int i = blockIdx.x * blockDim.x + threadIdx.x;
  if (i < n4) {
    float4 f = src[i];
    v4bf o;
    o[0] = (__bf16)f.x; o[1] = (__bf16)f.y;
    o[2] = (__bf16)f.z; o[3] = (__bf16)f.w;
    dst[i] = o;
  }
}

// ---------------------------------------------------------------------------
// Stage 1: QKV projection GEMM + RoPE + head scatter.
// The 64-wide N tile aligns with one head*section, so RoPE pairs (d, d+32)
// are accumulators (j, j+2).  Q scaled by 1/sqrt(64) on store.
// V stored transposed [B,H,64,T] so attention PV A-fragments are contiguous.
// ---------------------------------------------------------------------------
__global__ __launch_bounds__(256) void qkv_rope_kernel(
    const __bf16* __restrict__ xb, const __bf16* __restrict__ wb,
    const float* __restrict__ cosT, const float* __restrict__ sinT,
    __bf16* __restrict__ Qb, __bf16* __restrict__ Kb, __bf16* __restrict__ Vtb) {
  const int lane = threadIdx.x & 31;
  const int wid  = blockIdx.x * (blockDim.x >> 5) + (threadIdx.x >> 5);
  const int NT   = (3 * DMODEL) / 64;          // 48 N-tiles
  const int m0   = (wid / NT) * 16;
  const int n0   = (wid % NT) * 64;

  v8f acc[4] = {};
  gemm_16x64(xb + (size_t)m0 * DMODEL, wb + (size_t)n0 * DMODEL, lane, acc);

  const int n    = lane & 15;
  const int hi   = lane >> 4;
  const int sect = n0 / DMODEL;                // 0=q 1=k 2=v
  const int fin  = n0 % DMODEL;
  const int h    = fin / HDIM;

#pragma unroll
  for (int i = 0; i < 8; ++i) {
    const int m  = m0 + i + 8 * hi;            // global row (b*T + t)
    const int b_ = m >> 11;
    const int t  = m & (SEQ - 1);
    if (sect == 2) {                           // V -> transposed [B,H,64,T]
#pragma unroll
      for (int j = 0; j < 4; ++j) {
        const int d = j * 16 + n;
        Vtb[(((size_t)(b_ * NHEADS + h) * HDIM + d) << 11) + t] = (__bf16)acc[j][i];
      }
    } else {                                   // Q/K with RoPE
      const float* cr = cosT + (size_t)t * HDIM;
      const float* sr = sinT + (size_t)t * HDIM;
      const size_t base = ((size_t)(b_ * NHEADS + h) * SEQ + t) * HDIM;
#pragma unroll
      for (int j = 0; j < 2; ++j) {
        const int d1 = j * 16 + n;             // [0,32)
        const int d2 = d1 + 32;                // [32,64)
        const float u1 = acc[j][i], u2 = acc[j + 2][i];
        const float r1 = u1 * cr[d1] - u2 * sr[d1];
        const float r2 = u2 * cr[d2] + u1 * sr[d2];
        if (sect == 0) {
          Qb[base + d1] = (__bf16)(r1 * 0.125f);   // fold 1/sqrt(64)
          Qb[base + d2] = (__bf16)(r2 * 0.125f);
        } else {
          Kb[base + d1] = (__bf16)r1;
          Kb[base + d2] = (__bf16)r2;
        }
      }
    }
  }
}

// ---------------------------------------------------------------------------
// Stage 2: causal flash attention.  One wave = one (b,h,16-row q tile).
// S^T = K·Q^T (M=kv, N=q): softmax reduction over keys is intra-lane + one
// shfl_xor(16).  O^T = V^T·P^T with P^T assembled as a B-fragment via one
// half-wave exchange.  All 8 fragment loads of a kv-step issue before any
// WMMA so V loads hide behind S-WMMA + softmax VALU.
// ---------------------------------------------------------------------------
__global__ __launch_bounds__(256) void attn_kernel(
    const __bf16* __restrict__ Qb, const __bf16* __restrict__ Kb,
    const __bf16* __restrict__ Vtb, __bf16* __restrict__ Ob) {
  const int lane = threadIdx.x & 31;
  const int wid  = blockIdx.x * (blockDim.x >> 5) + (threadIdx.x >> 5);
  const int bh   = wid >> 7;                   // / (T/16)
  const int q0   = (wid & 127) << 4;
  const int b_   = bh >> 4;
  const int h    = bh & 15;
  const int n    = lane & 15;
  const int hi   = lane >> 4;

  const __bf16* Qh = Qb  + (size_t)bh * SEQ * HDIM;
  const __bf16* Kh = Kb  + (size_t)bh * SEQ * HDIM;
  const __bf16* Vh = Vtb + (size_t)bh * HDIM * SEQ;

  // loop-invariant Q^T B-fragments (K=c contiguous in Qh rows)
  const v16bf qt0 = load_bT(Qh + (size_t)q0 * HDIM + 0,  HDIM, lane);
  const v16bf qt1 = load_bT(Qh + (size_t)q0 * HDIM + 32, HDIM, lane);

  float m_run = -1e30f, l_run = 0.0f;
  v8f o[4] = {};
  const int qg   = q0 + n;                     // this lane's query index
  const int kend = q0 + 15;

  for (int kv = 0; kv <= kend; kv += 32) {
    // ---- issue all loads for this kv step up front ----
    v16bf ka0 = load_a(Kh + (size_t)kv * HDIM + 0,         HDIM, lane);
    v16bf ka1 = load_a(Kh + (size_t)kv * HDIM + 32,        HDIM, lane);
    v16bf ka2 = load_a(Kh + (size_t)(kv + 16) * HDIM + 0,  HDIM, lane);
    v16bf ka3 = load_a(Kh + (size_t)(kv + 16) * HDIM + 32, HDIM, lane);
    v16bf va[4];
#pragma unroll
    for (int j = 0; j < 4; ++j)
      va[j] = load_a(Vh + (size_t)(j * 16) * SEQ + kv, SEQ, lane);  // V^T rows = d

    if (kv + 32 <= kend) {                     // wave-uniform: prefetch next K tile
      __builtin_prefetch(Kh + (size_t)(kv + 32) * HDIM + (lane & 15) * HDIM + hi * 32, 0, 0);
    }

    v8f s0 = {}, s1 = {};
    s0 = WMMA_BF16(ka0, qt0, s0);
    s0 = WMMA_BF16(ka1, qt1, s0);
    s1 = WMMA_BF16(ka2, qt0, s1);
    s1 = WMMA_BF16(ka3, qt1, s1);

    if (kv + 16 > q0) {                        // diagonal tile (wave-uniform test)
#pragma unroll
      for (int i = 0; i < 8; ++i) {
        s0[i] = (kv + i + 8 * hi      > qg) ? -1e30f : s0[i];
        s1[i] = (kv + 16 + i + 8 * hi > qg) ? -1e30f : s1[i];
      }
    }

    float tm = -1e30f;
#pragma unroll
    for (int i = 0; i < 8; ++i) { tm = fmaxf(tm, s0[i]); tm = fmaxf(tm, s1[i]); }
    tm = fmaxf(tm, __shfl_xor(tm, 16, 32));
    const float m_new = fmaxf(m_run, tm);
    const float scale = __expf(m_run - m_new);

    float p0[8], p1[8], ls = 0.0f;
#pragma unroll
    for (int i = 0; i < 8; ++i) {
      p0[i] = __expf(s0[i] - m_new);
      p1[i] = __expf(s1[i] - m_new);
      ls += p0[i] + p1[i];
    }
    l_run = l_run * scale + (ls + __shfl_xor(ls, 16, 32));
    m_run = m_new;
#pragma unroll
    for (int j = 0; j < 4; ++j)
#pragma unroll
      for (int i = 0; i < 8; ++i) o[j][i] *= scale;

    // P^T B-fragment: lane needs kv tile #hi column q; own half + partner half.
    v16bf pf;
#pragma unroll
    for (int i = 0; i < 8; ++i) {
      const float x0 = __shfl_xor(p0[i], 16, 32);
      const float x1 = __shfl_xor(p1[i], 16, 32);
      pf[i]     = (__bf16)(hi ? x1 : p0[i]);   // K = hi*16 + i
      pf[8 + i] = (__bf16)(hi ? p1[i] : x0);   // K = hi*16 + 8 + i
    }

#pragma unroll
    for (int j = 0; j < 4; ++j)
      o[j] = WMMA_BF16(va[j], pf, o[j]);
  }

  const float inv_l = 1.0f / l_run;
  const size_t row = ((size_t)b_ * SEQ + q0 + n) * DMODEL + h * HDIM;
#pragma unroll
  for (int j = 0; j < 4; ++j)
#pragma unroll
    for (int i = 0; i < 8; ++i) {
      const int d = j * 16 + i + 8 * hi;
      Ob[row + d] = (__bf16)(o[j][i] * inv_l);
    }
}

// ---------------------------------------------------------------------------
// Stage 3: output projection, out[m,o] = sum_c O[m,c] * w_out[o,c], fp32 out.
// ---------------------------------------------------------------------------
__global__ __launch_bounds__(256) void outproj_kernel(
    const __bf16* __restrict__ Ob, const __bf16* __restrict__ wb,
    float* __restrict__ out) {
  const int lane = threadIdx.x & 31;
  const int wid  = blockIdx.x * (blockDim.x >> 5) + (threadIdx.x >> 5);
  const int NT   = DMODEL / 64;                // 16
  const int m0   = (wid / NT) * 16;
  const int n0   = (wid % NT) * 64;

  v8f acc[4] = {};
  gemm_16x64(Ob + (size_t)m0 * DMODEL, wb + (size_t)n0 * DMODEL, lane, acc);

  const int n = lane & 15, hi = lane >> 4;
#pragma unroll
  for (int i = 0; i < 8; ++i) {
    const int m = m0 + i + 8 * hi;
#pragma unroll
    for (int j = 0; j < 4; ++j)
      out[(size_t)m * DMODEL + n0 + j * 16 + n] = acc[j][i];
  }
}

// ---------------------------------------------------------------------------
extern "C" void kernel_launch(void* const* d_in, const int* in_sizes, int n_in,
                              void* d_out, int out_size, void* d_ws, size_t ws_size,
                              hipStream_t stream) {
  const float* x     = (const float*)d_in[0];   // (4,2048,1024)
  const float* cosT  = (const float*)d_in[1];   // (2048,64)
  const float* sinT  = (const float*)d_in[2];   // (2048,64)
  const float* w_qkv = (const float*)d_in[3];   // (3072,1024)
  const float* w_out = (const float*)d_in[4];   // (1024,1024)
  float* out = (float*)d_out;

  char* ws = (char*)d_ws;
  size_t off = 0;
  const size_t N_X    = (size_t)MROWS * DMODEL;          // 8,388,608
  const size_t N_WQKV = (size_t)3 * DMODEL * DMODEL;     // 3,145,728
  const size_t N_WOUT = (size_t)DMODEL * DMODEL;         // 1,048,576
  const size_t N_HEADV = (size_t)BATCH * NHEADS * SEQ * HDIM;  // 8,388,608

  __bf16* xb    = (__bf16*)(ws + off); off += N_X    * 2;
  __bf16* wqkvb = (__bf16*)(ws + off); off += N_WQKV * 2;
  __bf16* woutb = (__bf16*)(ws + off); off += N_WOUT * 2;
  __bf16* Qb    = (__bf16*)(ws + off); off += N_HEADV * 2;
  __bf16* Kb    = (__bf16*)(ws + off); off += N_HEADV * 2;
  __bf16* Vtb   = (__bf16*)(ws + off); off += N_HEADV * 2;
  __bf16* Ob    = (__bf16*)(ws + off); off += N_HEADV * 2;   // [B,T,1024]

  cvt_f32_to_bf16<<<(int)((N_X / 4    + 255) / 256), 256, 0, stream>>>(
      (const float4*)x,     (v4bf*)xb,    (int)(N_X / 4));
  cvt_f32_to_bf16<<<(int)((N_WQKV / 4 + 255) / 256), 256, 0, stream>>>(
      (const float4*)w_qkv, (v4bf*)wqkvb, (int)(N_WQKV / 4));
  cvt_f32_to_bf16<<<(int)((N_WOUT / 4 + 255) / 256), 256, 0, stream>>>(
      (const float4*)w_out, (v4bf*)woutb, (int)(N_WOUT / 4));

  // 512 M-tiles x 48 N-tiles = 24576 waves / 8 waves per block
  qkv_rope_kernel<<<3072, 256, 0, stream>>>(xb, wqkvb, cosT, sinT, Qb, Kb, Vtb);
  // 4*16*128 q-tiles = 8192 waves
  attn_kernel<<<1024, 256, 0, stream>>>(Qb, Kb, Vtb, Ob);
  // 512 x 16 = 8192 waves
  outproj_kernel<<<1024, 256, 0, stream>>>(Ob, woutb, out);
}
